// Coder_5609227288756
// MI455X (gfx1250) — matrix operand order repaired and checked by reading, not compiled, not run on hardware
//
#include <hip/hip_runtime.h>

// ---------------------------------------------------------------------------
// MI455X / gfx1250 implementation of the attention+FF block.
// All heavy math runs on v_wmma_f32_16x16x32_bf16 (fp32 accumulate).
// Flash-attention style: the [B*H, S, S] score tensor is never materialized.
// The big E x E GEMMs stage their B-tiles in LDS via the gfx1250 async
// global->LDS path (ASYNCcnt) when the toolchain exposes the builtin.
// ---------------------------------------------------------------------------

#define DEV __device__ __forceinline__

typedef unsigned short u16;
typedef __attribute__((ext_vector_type(16))) __bf16 v16bf;
typedef __attribute__((ext_vector_type(8)))  float  v8f;

constexpr int CB  = 4;     // batch
constexpr int CS  = 2048;  // sequence
constexpr int CE  = 1024;  // embed
constexpr int CH  = 16;    // heads
constexpr int CHF = 64;    // head feature

union FragB { v16bf v; uint4 q[2]; };

// ---- gfx1250 async global->LDS copy (guarded; sync fallback) --------------
#if defined(__has_builtin)
#  if __has_builtin(__builtin_amdgcn_global_load_async_to_lds_b128)
#    define HAVE_ASYNC_LDS 1
#  endif
#endif
#ifndef HAVE_ASYNC_LDS
#  define HAVE_ASYNC_LDS 0
#endif

// Builtin signature (from compiler diagnostic): arg0 is
// 'int __attribute__((vector_size(16))) __device__ *' i.e. v4i in AS1.
typedef int v4i_ __attribute__((vector_size(16)));
typedef __attribute__((address_space(1))) v4i_* g_v4i_ptr;
typedef __attribute__((address_space(3))) v4i_* l_v4i_ptr;

DEV void copy16_g2l(u16* lds_dst, const u16* gsrc) {
#if HAVE_ASYNC_LDS
  __builtin_amdgcn_global_load_async_to_lds_b128(
      (g_v4i_ptr)(gsrc), (l_v4i_ptr)(lds_dst), /*offset=*/0, /*cpol=*/0);
#else
  *(uint4*)lds_dst = *(const uint4*)gsrc;
#endif
}

DEV void async_copies_wait() {
#if HAVE_ASYNC_LDS
#  if __has_builtin(__builtin_amdgcn_s_wait_asynccnt)
  __builtin_amdgcn_s_wait_asynccnt(0);
#  else
  asm volatile("s_wait_asynccnt 0" ::: "memory");
#  endif
#endif
}

DEV u16 f2bf(float f) {
  unsigned u = __float_as_uint(f);
  u += 0x7FFFu + ((u >> 16) & 1u);   // round-to-nearest-even
  return (u16)(u >> 16);
}

// A fragment: 16x32 bf16, source row-major, leading dim ld.
// ISA layout: lane m (0..15) / m+16 hold K = {h*8+0..7, 16+h*8+0..7} of row m.
DEV v16bf load_afrag(const u16* base, int ld, int kbase) {
  const int lane = threadIdx.x & 31;
  const int m = lane & 15, h = lane >> 4;
  FragB f;
  const u16* p = base + (long)m * ld + kbase + h * 8;
  f.q[0] = *(const uint4*)(p);
  f.q[1] = *(const uint4*)(p + 16);
  return f.v;
}

// B fragment: 32x16 bf16 given as B^T row-major [16(N), K], leading dim ld.
// ISA layout: lane n (0..15) / n+16 hold K = {h*8+0..7, 16+h*8+0..7} of column n.
DEV v16bf load_bfrag(const u16* bt, int ld, int kbase) {
  const int lane = threadIdx.x & 31;
  const int n = lane & 15, h = lane >> 4;
  FragB f;
  const u16* p = bt + (long)n * ld + kbase + h * 8;
  f.q[0] = *(const uint4*)(p);
  f.q[1] = *(const uint4*)(p + 16);
  return f.v;
}

DEV v8f wmma_bf16(v16bf a, v16bf b, v8f c) {
  return __builtin_amdgcn_wmma_f32_16x16x32_bf16(
      false, a, false, b, (short)0, c, false, false);
}

// ---------------------------------------------------------------------------
__global__ void k_cvt(const float* __restrict__ in, u16* __restrict__ out, long n) {
  long i = (long)blockIdx.x * blockDim.x + threadIdx.x;
  if (i < n) out[i] = f2bf(in[i]);
}

// Transposing convert: in [D0, D1, D2] fp32 -> out [D0, D2, D1] bf16
__global__ void k_cvt_t(const float* __restrict__ in, u16* __restrict__ out,
                        int D0, int D1, int D2) {
  long i = (long)blockIdx.x * blockDim.x + threadIdx.x;
  long n = (long)D0 * D1 * D2;
  if (i >= n) return;
  int j  = (int)(i % D1);
  long t = i / D1;
  int jj = (int)(t % D2);
  int b  = (int)(t / D2);
  out[i] = f2bf(in[((long)b * D1 + j) * D2 + jj]);
}

// ---------------------------------------------------------------------------
// QKV projection: one wave per (b, h, 16-row tile). A-frag reused 12x.
__global__ __launch_bounds__(32)
void k_qkv(const u16* __restrict__ xb,
           const u16* __restrict__ Wqt, const u16* __restrict__ Wkt,
           const u16* __restrict__ Wvt,
           u16* __restrict__ q, u16* __restrict__ k, u16* __restrict__ vt) {
  const int s0 = blockIdx.x * 16;
  const int h  = blockIdx.y;
  const int b  = blockIdx.z;
  const int lane = threadIdx.x & 31;
  const int lm = lane & 15, lh = lane >> 4;

  const u16* A  = xb + ((long)b * CS + s0) * CE;
  const u16* Bq = Wqt + (long)h * CHF * CE;
  const u16* Bk = Wkt + (long)h * CHF * CE;
  const u16* Bv = Wvt + (long)h * CHF * CE;

  v8f accq[4] = {}, acck[4] = {}, accv[4] = {};

  for (int kk = 0; kk < CE; kk += 32) {
    v16bf a = load_afrag(A, CE, kk);
#pragma unroll
    for (int t = 0; t < 4; ++t) {
      accq[t] = wmma_bf16(a, load_bfrag(Bq + (long)t * 16 * CE, CE, kk), accq[t]);
      acck[t] = wmma_bf16(a, load_bfrag(Bk + (long)t * 16 * CE, CE, kk), acck[t]);
      accv[t] = wmma_bf16(a, load_bfrag(Bv + (long)t * 16 * CE, CE, kk), accv[t]);
    }
  }

  const int z = b * CH + h;
#pragma unroll
  for (int t = 0; t < 4; ++t)
#pragma unroll
    for (int r = 0; r < 8; ++r) {
      const int row = s0 + r + lh * 8;
      const int f   = t * 16 + lm;
      q[((long)z * CS + row) * CHF + f]  = f2bf(accq[t][r]);
      k[((long)z * CS + row) * CHF + f]  = f2bf(acck[t][r]);
      vt[((long)z * CHF + f) * CS + row] = f2bf(accv[t][r]);
    }
}

// ---------------------------------------------------------------------------
// Flash attention: one wave per (b, h, 16-query tile); online softmax.
__global__ __launch_bounds__(32)
void k_attn(const u16* __restrict__ q, const u16* __restrict__ kmat,
            const u16* __restrict__ vt, const int* __restrict__ mask,
            u16* __restrict__ zb) {
  __shared__ __align__(16) u16 ldsP[16 * 32];

  const int s0 = blockIdx.x * 16;
  const int h  = blockIdx.y;
  const int b  = blockIdx.z;
  const int z  = b * CH + h;
  const int lane = threadIdx.x & 31;
  const int lm = lane & 15, lh = lane >> 4;

  const u16* qz = q    + ((long)z * CS + s0) * CHF;
  const u16* kz = kmat + (long)z * CS * CHF;
  const u16* vz = vt   + (long)z * CHF * CS;

  const v16bf aq0 = load_afrag(qz, CHF, 0);
  const v16bf aq1 = load_afrag(qz, CHF, 32);

  float mrow[8], lrow[8];
#pragma unroll
  for (int r = 0; r < 8; ++r) { mrow[r] = -1e30f; lrow[r] = 0.f; }
  v8f acc[4] = {};

  for (int kt = 0; kt < CS; kt += 32) {
    v8f s0f = {}, s1f = {};
    s0f = wmma_bf16(aq0, load_bfrag(kz + (long)kt * CHF, CHF, 0),  s0f);
    s0f = wmma_bf16(aq1, load_bfrag(kz + (long)kt * CHF, CHF, 32), s0f);
    s1f = wmma_bf16(aq0, load_bfrag(kz + (long)(kt + 16) * CHF, CHF, 0),  s1f);
    s1f = wmma_bf16(aq1, load_bfrag(kz + (long)(kt + 16) * CHF, CHF, 32), s1f);

    float a_scale[8];
#pragma unroll
    for (int r = 0; r < 8; ++r) {
      const int row = s0 + r + lh * 8;
      const int c0  = kt + lm;
      float v0 = s0f[r], v1 = s1f[r];
      if (mask[(long)row * CS + c0] == 0)      v0 = -1e9f;  // mask before scale
      if (mask[(long)row * CS + c0 + 16] == 0) v1 = -1e9f;
      v0 *= 0.125f; v1 *= 0.125f;                           // 1/sqrt(HF)
      float cm = fmaxf(v0, v1);
#pragma unroll
      for (int off = 1; off < 16; off <<= 1)
        cm = fmaxf(cm, __shfl_xor(cm, off, 32));
      const float mnew  = fmaxf(mrow[r], cm);
      const float alpha = __expf(mrow[r] - mnew);
      const float p0 = __expf(v0 - mnew);
      const float p1 = __expf(v1 - mnew);
      float rs = p0 + p1;
#pragma unroll
      for (int off = 1; off < 16; off <<= 1)
        rs += __shfl_xor(rs, off, 32);
      lrow[r] = lrow[r] * alpha + rs;
      mrow[r] = mnew;
      a_scale[r] = alpha;
      const int lbase = (r + lh * 8) * 32;
      ldsP[lbase + lm]      = f2bf(p0);
      ldsP[lbase + lm + 16] = f2bf(p1);
    }

#pragma unroll
    for (int t = 0; t < 4; ++t)
#pragma unroll
      for (int r = 0; r < 8; ++r)
        acc[t][r] *= a_scale[r];

    FragB fp;
    fp.q[0] = *(const uint4*)(ldsP + lm * 32 + lh * 8);
    fp.q[1] = *(const uint4*)(ldsP + lm * 32 + lh * 8 + 16);
    const v16bf ap = fp.v;

#pragma unroll
    for (int t = 0; t < 4; ++t)
      acc[t] = wmma_bf16(ap, load_bfrag(vz + (long)t * 16 * CS, CS, kt), acc[t]);
  }

#pragma unroll
  for (int t = 0; t < 4; ++t)
#pragma unroll
    for (int r = 0; r < 8; ++r) {
      const int row = s0 + r + lh * 8;
      const int e   = h * CHF + t * 16 + lm;
      zb[((long)b * CS + row) * CE + e] = f2bf(acc[t][r] / lrow[r]);
    }
}

// ---------------------------------------------------------------------------
// Generic bf16 GEMM with LDS-staged B tile (shared by all 4 waves).
__global__ __launch_bounds__(128)
void k_gemm(const u16* __restrict__ A, const u16* __restrict__ Bt,
            u16* __restrict__ Cbf, float* __restrict__ Cf,
            const float* __restrict__ bias,
            int M, int N, int K,
            long strideA, long strideB, long strideC, int relu) {
  __shared__ __align__(16) u16 ldsB[64 * 64];   // [n(64)][k(64)], 8 KB
  const int b    = blockIdx.z;
  const int wave = threadIdx.x >> 5;
  const int lane = threadIdx.x & 31;
  const int lm = lane & 15, lh = lane >> 4;
  const int row0 = blockIdx.y * 64 + wave * 16;
  const int col0 = blockIdx.x * 64;

  const u16* Ab = A  + (long)b * strideA + (long)row0 * K;
  const u16* Bb = Bt + (long)b * strideB + (long)col0 * K;

  v8f acc[4] = {};
  for (int kk = 0; kk < K; kk += 64) {
    // cooperative B stage: 64 rows x 128 B = 512 x 16 B chunks, async->LDS
#pragma unroll
    for (int i = 0; i < 4; ++i) {
      const int chunk = threadIdx.x + i * 128;   // 0..511
      const int n = chunk >> 3;                  // B^T row
      const int c = (chunk & 7) * 8;             // k offset (u16 units)
      copy16_g2l(&ldsB[n * 64 + c], Bb + (long)n * K + kk + c);
    }
    async_copies_wait();
    __syncthreads();

    if (kk + 64 < K)  // stream A ahead -> global_prefetch_b8
      __builtin_prefetch(Ab + (long)lm * K + kk + 64, 0, 0);

#pragma unroll
    for (int ks = 0; ks < 64; ks += 32) {
      v16bf a = load_afrag(Ab, K, kk + ks);
#pragma unroll
      for (int t = 0; t < 4; ++t)
        acc[t] = wmma_bf16(a, load_bfrag(ldsB + t * 16 * 64, 64, ks), acc[t]);
    }
    __syncthreads();
  }

#pragma unroll
  for (int t = 0; t < 4; ++t)
#pragma unroll
    for (int r = 0; r < 8; ++r) {
      const int row = row0 + r + lh * 8;
      const int col = col0 + t * 16 + lm;
      const long idx = (long)b * strideC + (long)row * N + col;
      float v = acc[t][r];
      if (bias) v += bias[col];
      if (relu) v = fmaxf(v, 0.f);
      if (Cf) Cf[idx] = v; else Cbf[idx] = f2bf(v);
    }
}

// ---------------------------------------------------------------------------
extern "C" void kernel_launch(void* const* d_in, const int* in_sizes, int n_in,
                              void* d_out, int out_size, void* d_ws, size_t ws_size,
                              hipStream_t stream) {
  (void)in_sizes; (void)n_in; (void)out_size; (void)ws_size;
  const float* x    = (const float*)d_in[0];
  const int*   mask = (const int*)d_in[1];
  const float* Wq   = (const float*)d_in[2];
  const float* Wk   = (const float*)d_in[3];
  const float* Wv   = (const float*)d_in[4];
  const float* Wr   = (const float*)d_in[5];
  const float* Wff  = (const float*)d_in[6];
  const float* bff  = (const float*)d_in[7];
  float* out = (float*)d_out;

  const long nBSE = (long)CB * CS * CE;
  const long nW   = (long)CH * CE * CHF;     // == E*E
  const long nWr  = (long)CB * CE * CE;
  const long nQ   = (long)CB * CH * CS * CHF;

  u16* p = (u16*)d_ws;
  u16* xb  = p; p += nBSE;
  u16* Wqt = p; p += nW;
  u16* Wkt = p; p += nW;
  u16* Wvt = p; p += nW;
  u16* Wrt = p; p += nWr;
  u16* Wfb = p; p += nW;
  u16* qb  = p; p += nQ;
  u16* kb  = p; p += nQ;
  u16* vtb = p; p += nQ;
  u16* zb  = p; p += nBSE;
  u16* tb  = p; p += nBSE;

  k_cvt  <<<dim3((unsigned)((nBSE + 255) / 256)), 256, 0, stream>>>(x,   xb,  nBSE);
  k_cvt  <<<dim3((unsigned)((nW   + 255) / 256)), 256, 0, stream>>>(Wff, Wfb, nW);
  k_cvt_t<<<dim3((unsigned)((nW   + 255) / 256)), 256, 0, stream>>>(Wq, Wqt, CH, CE, CHF);
  k_cvt_t<<<dim3((unsigned)((nW   + 255) / 256)), 256, 0, stream>>>(Wk, Wkt, CH, CE, CHF);
  k_cvt_t<<<dim3((unsigned)((nW   + 255) / 256)), 256, 0, stream>>>(Wv, Wvt, CH, CE, CHF);
  k_cvt_t<<<dim3((unsigned)((nWr  + 255) / 256)), 256, 0, stream>>>(Wr, Wrt, CB, CE, CE);

  k_qkv<<<dim3(CS / 16, CH, CB), 32, 0, stream>>>(xb, Wqt, Wkt, Wvt, qb, kb, vtb);
  k_attn<<<dim3(CS / 16, CH, CB), 32, 0, stream>>>(qb, kb, vtb, mask, zb);

  k_gemm<<<dim3(CE / 64, CS / 64, CB), 128, 0, stream>>>(
      zb, Wrt, tb, nullptr, nullptr,
      CS, CE, CE, (long)CS * CE, (long)CE * CE, (long)CS * CE, 0);

  k_gemm<<<dim3(CE / 64, CS / 64, CB), 128, 0, stream>>>(
      tb, Wfb, nullptr, out, bff,
      CS, CE, CE, (long)CS * CE, 0, (long)CS * CE, 1);
}
